// QuantLinearORT_67130338836764
// MI455X (gfx1250) — matrix operand order repaired and checked by reading, not compiled, not run on hardware
//
#include <hip/hip_runtime.h>
#include <hip/hip_bf16.h>
#include <stdint.h>

typedef __attribute__((ext_vector_type(16))) _Float16 v16h;
typedef __attribute__((ext_vector_type(8)))  _Float16 v8h;
typedef __attribute__((ext_vector_type(8)))  float    v8f;
typedef __attribute__((ext_vector_type(4)))  float    v4f;
typedef __attribute__((ext_vector_type(4)))  int      v4i;

#define KF   4096              // IN_FEATURES
#define NF   11008             // OUT_FEATURES
#define MF   512               // M_TOKENS
#define GS   128               // group size
#define KB   (KF / GS)         // 32 k-blocks
#define MT   128               // M rows per workgroup
#define NT   128               // N cols per workgroup (8 waves x 16)
#define AROW 136               // padded halfs per LDS row (272 B = 17 x 16 B)

__global__ __launch_bounds__(256)
void q4_wmma_gemm(const float* __restrict__ x,
                  const int*   __restrict__ qw,
                  const float* __restrict__ scales,
                  const int*   __restrict__ qzeros,
                  float*       __restrict__ out)
{
    __shared__ _Float16 As[MT * AROW];   // 34816 B of the 320 KB WGP LDS

    const int tid   = threadIdx.x;
    const int wave  = tid >> 5;
    const int lane  = tid & 31;
    const int laneN = lane & 15;         // column / row within 16
    const int laneH = lane >> 4;         // K half-select

    const int mBase = blockIdx.y * MT;
    const int nCol  = blockIdx.x * NT + wave * 16 + laneN;

    // cooperative A staging coords: 128 rows x 128 cols per kb, 64 f32/thread
    const int sRow  = tid >> 1;
    const int sColB = (tid & 1) * 64;
    const float* xRow = x + (size_t)(mBase + sRow) * KF + sColB;
    _Float16*    aDst = &As[sRow * AROW + sColB];

    const int* qRow = qw + (size_t)nCol * (KB * (GS / 2));  // 2048 int32 per n

    v8f acc[8] = {};

    for (int kb = 0; kb < KB; ++kb) {
        // per-(n, group) scale / zero-point
        const float s  = scales[nCol * KB + kb];
        const int   zb = qzeros[(nCol * KB + kb) >> 1];
        const int   z  = (kb & 1) ? ((zb >> 4) & 15) : (zb & 15);
        const float zs = (float)z * s;
        const int*  qg = qRow + kb * (GS / 2);
        if (kb + 1 < KB) {
            __builtin_prefetch(qg + (GS / 2), 0, 0);
            __builtin_prefetch(xRow + (kb + 1) * GS, 0, 0);
        }

        // ---- stage whole group: 128x128 f32 -> f16 -> LDS, one barrier pair ----
        {
            const float* p = xRow + kb * GS;
            #pragma unroll
            for (int j = 0; j < 4; ++j) {           // 16 cols per step
                v4f f0 = *(const v4f*)(p + j * 16);
                v4f f1 = *(const v4f*)(p + j * 16 + 4);
                v4f f2 = *(const v4f*)(p + j * 16 + 8);
                v4f f3 = *(const v4f*)(p + j * 16 + 12);
                v8h h0, h1;
                #pragma unroll
                for (int i = 0; i < 4; ++i) {
                    h0[i]     = (_Float16)f0[i];
                    h0[4 + i] = (_Float16)f1[i];
                    h1[i]     = (_Float16)f2[i];
                    h1[4 + i] = (_Float16)f3[i];
                }
                *(v8h*)(aDst + j * 16)     = h0;
                *(v8h*)(aDst + j * 16 + 8) = h1;
            }
        }
        __syncthreads();

        // ---- 4 chunks of K=32, no fences between: dequant + 8 WMMA each ----
        #pragma unroll
        for (int c = 0; c < 4; ++c) {
            // B fragment: element e <-> K = c*32 + laneH*16 + e (lane = column n)
            v16h bf;
            {
                const int* qc = qg + c * 16 + laneH * 8;   // 8 ints = 16 k values
                v4i w0 = *(const v4i*)qc;
                v4i w1 = *(const v4i*)(qc + 4);
                #pragma unroll
                for (int i = 0; i < 4; ++i) {
                    bf[2 * i]         = (_Float16)((float)( w0[i]        & 15) * s - zs);
                    bf[2 * i + 1]     = (_Float16)((float)((w0[i] >> 4)  & 15) * s - zs);
                    bf[8 + 2 * i]     = (_Float16)((float)( w1[i]        & 15) * s - zs);
                    bf[8 + 2 * i + 1] = (_Float16)((float)((w1[i] >> 4)  & 15) * s - zs);
                }
            }

            // A fragments: lane holds row m; lanes0-15 K 0-15, lanes16-31 K 16-31
            #pragma unroll
            for (int ms = 0; ms < 8; ++ms) {
                const _Float16* ap = &As[(ms * 16 + laneN) * AROW + c * 32 + laneH * 16];
                v8h alo = *(const v8h*)(ap);        // K base .. base+7
                v8h ahi = *(const v8h*)(ap + 8);    // K base+8 .. base+15
                v16h af;
                #pragma unroll
                for (int i = 0; i < 8; ++i) { af[i] = alo[i]; af[8 + i] = ahi[i]; }
                acc[ms] = __builtin_amdgcn_wmma_f32_16x16x32_f16(
                    /*neg_a=*/false, af, /*neg_b=*/false, bf,
                    /*c_mod=*/(short)0, acc[ms],
                    /*reuse_a=*/false, /*reuse_b=*/false);
            }
        }
        __syncthreads();
    }

    // ---- store C: VGPR r <-> row r + laneH*8, col = laneN ----
    #pragma unroll
    for (int ms = 0; ms < 8; ++ms) {
        float* dst = out + (size_t)(mBase + ms * 16 + laneH * 8) * NF + nCol;
        #pragma unroll
        for (int r = 0; r < 8; ++r)
            dst[(size_t)r * NF] = acc[ms][r];
    }
}

extern "C" void kernel_launch(void* const* d_in, const int* in_sizes, int n_in,
                              void* d_out, int out_size, void* d_ws, size_t ws_size,
                              hipStream_t stream) {
    (void)in_sizes; (void)n_in; (void)d_ws; (void)ws_size; (void)out_size;
    const float* x      = (const float*)d_in[0];
    const int*   qw     = (const int*)d_in[1];
    const float* scales = (const float*)d_in[2];
    const int*   qzeros = (const int*)d_in[3];
    float*       out    = (float*)d_out;

    dim3 grid(NF / NT, MF / MT);   // (86, 4)
    dim3 block(256);               // 8 wave32s per workgroup
    q4_wmma_gemm<<<grid, block, 0, stream>>>(x, qw, scales, qzeros, out);
}
